// RefiningEncoderBlock_90683939487948
// MI455X (gfx1250) — compile-verified, matrix operand order checked by reading, not run on hardware
//
#include <hip/hip_runtime.h>
#include <hip/hip_bf16.h>
#include <math.h>
#include <stdint.h>

// ---------------------------------------------------------------------------
// RefiningEncoderBlock for MI455X (gfx1250, wave32, WMMA, bf16 data path,
// TDM-staged GEMM tiles).
//
// Algebraic fusion: window K/V rows == rows of (xn @ w_kv); the global
// token's K/V is one extra row.  One GEMM over (B*L + B) = 36880 rows gives
// kv (windows), kv_g (global attn) and kv_avg at once; same for q.
//
// MI455X reasoning: ~250 GFLOP vs 23.3 TB/s HBM -> bandwidth bound.  All
// GEMM operands live in bf16; weights pre-transposed to N x K bf16.  GEMM
// tiles are moved global->LDS by the Tensor Data Mover (one descriptor per
// tile, double buffered, TENSORcnt-tracked) so the waves only do
// ds_load_b128 + v_wmma_f32_16x16x32_bf16.
// ---------------------------------------------------------------------------

#define DIMC   512
#define NHEAD  16
#define HEADD  32
#define BATCH  16
#define LTOK   2304                   // 48*48
#define NROW   (BATCH*LTOK + BATCH)   // 36880 fused token rows
#define SCALEQ 0.17677669529663689f   // 32^-0.5

typedef __attribute__((ext_vector_type(16))) __bf16 v16bf;
typedef __attribute__((ext_vector_type(8)))  __bf16 v8bf;
typedef __attribute__((ext_vector_type(8)))  float  v8f;
typedef __attribute__((ext_vector_type(4)))  unsigned uint4v;
typedef __attribute__((ext_vector_type(8)))  unsigned uint8v;

union AFrag {
  v16bf v;
  v8bf h[2];
  __bf16 e[16];
  unsigned u[8];
};

union F8 {
  v8bf v;
  __bf16 e[8];
};

__device__ inline void zero_frag(AFrag& f) {
#pragma unroll
  for (int i = 0; i < 8; ++i) f.u[i] = 0u;
}

// ---------------------------------------------------------------------------
// Tensor Data Mover: 2D bf16 tile (tileX x tileY elements, X contiguous)
// from global (row stride strideX elements) into LDS at ldsOff, packed
// row-major.  Rows/cols beyond tensorX/tensorY zero-fill (OOB rule), which
// also implements the M-edge guard for free.  Descriptor per ISA 8.3/8.4:
//   group0: count=1 | lds_addr | global_addr | type=2
//   group1: data_size=2B, tensor dims, tile dims, dim0 stride
// Issued by one wave; tracked by TENSORcnt.
// ---------------------------------------------------------------------------
__device__ inline void tdm_load_2d(unsigned ldsOff, const void* gptr,
                                   unsigned tileX, unsigned tileY,
                                   unsigned tensorX, unsigned tensorY,
                                   unsigned strideX) {
  unsigned long long ga = (unsigned long long)(uintptr_t)gptr;
  uint4v g0;
  g0[0] = 1u;                                  // count=1, user mode
  g0[1] = ldsOff;                              // lds_addr (bytes)
  g0[2] = (unsigned)ga;                        // global_addr[31:0]
  g0[3] = (unsigned)(ga >> 32) | (2u << 30);   // global_addr[56:32] | type=2
  uint8v g1;
  g1[0] = 1u << 16;                            // data_size=1 -> 2 bytes
  g1[1] = (tensorX & 0xFFFFu) << 16;           // tensor_dim0[15:0]
  g1[2] = (tensorX >> 16) | ((tensorY & 0xFFFFu) << 16);
  g1[3] = (tensorY >> 16) | (tileX << 16);     // tensor_dim1[31:16] | tile_dim0
  g1[4] = tileY & 0xFFFFu;                     // tile_dim1 (tile_dim2=0 -> 2D)
  g1[5] = strideX;                             // tensor_dim0_stride[31:0]
  g1[6] = 0u;
  g1[7] = 0u;
  asm volatile("tensor_load_to_lds %0, %1" : : "s"(g0), "s"(g1) : "memory");
}

// ---------------- relative position index table (compile-time) -------------
struct RelIdx { int v[36 * 37]; };
constexpr RelIdx make_rel_idx() {
  RelIdx r{};
  for (int p = 0; p < 36; ++p) {
    int x0p = p % 6, x1p = p / 6;
    for (int q = 0; q < 37; ++q) {
      int x0q = (q < 36) ? q % 6 : 0;
      int x1q = (q < 36) ? q / 6 : 0;
      r.v[p * 37 + q] = (x0p - x0q + 5) * 11 + (x1p - x1q + 5);
    }
  }
  return r;
}
__constant__ RelIdx c_rel = make_rel_idx();

__device__ inline float gelu_exact(float x) {
  return 0.5f * x * (1.0f + erff(x * 0.70710678118654752f));
}

// ---------------------------------------------------------------------------
// Weight transpose + f32->bf16:  S[K][N] -> D[N][K].
// ---------------------------------------------------------------------------
__global__ __launch_bounds__(256) void transpose_cvt(
    const float* __restrict__ S, __bf16* __restrict__ D, int K, int N) {
  __shared__ float t[32][33];
  int k0 = blockIdx.y * 32, n0 = blockIdx.x * 32;
  int tx = threadIdx.x & 31, ty = threadIdx.x >> 5;  // ty in [0,8)
#pragma unroll
  for (int i = 0; i < 32; i += 8)
    t[ty + i][tx] = S[(size_t)(k0 + ty + i) * N + n0 + tx];
  __syncthreads();
#pragma unroll
  for (int i = 0; i < 32; i += 8)
    D[(size_t)(n0 + ty + i) * K + k0 + tx] = (__bf16)t[tx][ty + i];
}

// ---------------------------------------------------------------------------
// LayerNorm over last dim (512), f32 in -> bf16 out.
// ---------------------------------------------------------------------------
__global__ __launch_bounds__(128) void ln_kernel(
    const float* __restrict__ src0, int R0, const float* __restrict__ src1,
    const float* __restrict__ g, const float* __restrict__ be,
    __bf16* __restrict__ dst) {
  __shared__ float red[4];
  int row = blockIdx.x;
  const float* src = (row < R0) ? (src0 + (size_t)row * DIMC)
                                : (src1 + (size_t)(row - R0) * DIMC);
  int tid = threadIdx.x;
  float v[4];
  float s = 0.f;
#pragma unroll
  for (int i = 0; i < 4; ++i) { v[i] = src[tid + i * 128]; s += v[i]; }
  for (int o = 16; o >= 1; o >>= 1) s += __shfl_xor(s, o, 32);
  if ((tid & 31) == 0) red[tid >> 5] = s;
  __syncthreads();
  float mean = (red[0] + red[1] + red[2] + red[3]) * (1.f / 512.f);
  __syncthreads();
  float q = 0.f;
#pragma unroll
  for (int i = 0; i < 4; ++i) { float d = v[i] - mean; q += d * d; }
  for (int o = 16; o >= 1; o >>= 1) q += __shfl_xor(q, o, 32);
  if ((tid & 31) == 0) red[tid >> 5] = q;
  __syncthreads();
  float var = (red[0] + red[1] + red[2] + red[3]) * (1.f / 512.f);
  float rstd = rsqrtf(var + 1e-3f);
  __bf16* out = dst + (size_t)row * DIMC;
#pragma unroll
  for (int i = 0; i < 4; ++i) {
    int c = tid + i * 128;
    out[c] = (__bf16)((v[i] - mean) * rstd * g[c] + be[c]);
  }
}

// ---------------------------------------------------------------------------
// bf16 WMMA GEMM:  C = act(A[M,K] @ BT[N,K]^T + bias) (+res)
// Block tile 128x64, 8 waves (4x2), wave tile 32x32 (2x2 WMMA), K step 32.
// Tiles staged by TDM (wave0: A, wave1: B), double buffered; fragments load
// as 2x ds_load_b128 per lane (ISA 7.12.2 K-interleave by half-wave).
// ---------------------------------------------------------------------------
__global__ __launch_bounds__(256) void gemm_bf16_wmma(
    const __bf16* __restrict__ A, const __bf16* __restrict__ BT,
    const float* __restrict__ bias, const float* __restrict__ res,
    float* __restrict__ Cf, __bf16* __restrict__ Cb,
    int M, int N, int K, int act) {
  __shared__ alignas(16) __bf16 As[2][128][32];
  __shared__ alignas(16) __bf16 Bs[2][64][32];
  int tid  = threadIdx.x;
  int lane = tid & 31, wave = tid >> 5;
  int wm = wave >> 1, wn = wave & 1;
  int m0 = blockIdx.x * 128, n0 = blockIdx.y * 64;
  int half = lane >> 4, lidx = lane & 15;

  const unsigned aOff0 = (unsigned)(uintptr_t)&As[0][0][0];
  const unsigned aOff1 = (unsigned)(uintptr_t)&As[1][0][0];
  const unsigned bOff0 = (unsigned)(uintptr_t)&Bs[0][0][0];
  const unsigned bOff1 = (unsigned)(uintptr_t)&Bs[1][0][0];
  const unsigned mRem = (unsigned)(M - m0);   // TDM OOB rows zero-fill
  const unsigned nRem = (unsigned)(N - n0);

  // prologue: DMA first tiles into buffer 0
  if (wave == 0)
    tdm_load_2d(aOff0, A + (size_t)m0 * K, 32, 128, (unsigned)K, mRem,
                (unsigned)K);
  else if (wave == 1)
    tdm_load_2d(bOff0, BT + (size_t)n0 * K, 32, 64, (unsigned)K, nRem,
                (unsigned)K);

  v8f acc[2][2] = {};
  int buf = 0;
  for (int k0 = 0; k0 < K; k0 += 32, buf ^= 1) {
    // free for waves with TENSORcnt==0; avoids an exec-mask dance
    __builtin_amdgcn_s_wait_tensorcnt(0);
    __syncthreads();
    // overlap: DMA next K tile into the other buffer while computing
    if (k0 + 32 < K) {
      if (wave == 0)
        tdm_load_2d(buf ? aOff0 : aOff1, A + (size_t)m0 * K + (k0 + 32),
                    32, 128, (unsigned)(K - (k0 + 32)), mRem, (unsigned)K);
      else if (wave == 1)
        tdm_load_2d(buf ? bOff0 : bOff1, BT + (size_t)n0 * K + (k0 + 32),
                    32, 64, (unsigned)(K - (k0 + 32)), nRem, (unsigned)K);
    }

    AFrag a[2], b[2];
#pragma unroll
    for (int t = 0; t < 2; ++t) {
      int rm = wm * 32 + t * 16 + lidx;
      a[t].h[0] = *(const v8bf*)&As[buf][rm][half * 8];
      a[t].h[1] = *(const v8bf*)&As[buf][rm][16 + half * 8];
      int rn = wn * 32 + t * 16 + lidx;
      b[t].h[0] = *(const v8bf*)&Bs[buf][rn][half * 8];
      b[t].h[1] = *(const v8bf*)&Bs[buf][rn][16 + half * 8];
    }
#pragma unroll
    for (int tm = 0; tm < 2; ++tm)
#pragma unroll
      for (int tn = 0; tn < 2; ++tn)
        acc[tm][tn] = __builtin_amdgcn_wmma_f32_16x16x32_bf16(
            false, a[tm].v, false, b[tn].v, (short)0, acc[tm][tn],
            false, false);
  }

  // epilogue: C/D layout -> VGPR r holds row (r + 8*half), lane -> col
#pragma unroll
  for (int tm = 0; tm < 2; ++tm)
#pragma unroll
    for (int tn = 0; tn < 2; ++tn)
#pragma unroll
      for (int r = 0; r < 8; ++r) {
        int row = m0 + wm * 32 + tm * 16 + r + half * 8;
        int col = n0 + wn * 32 + tn * 16 + lidx;
        if (row < M) {
          float v = acc[tm][tn][r] + bias[col];
          if (act) v = gelu_exact(v);
          if (res) v += res[(size_t)row * N + col];
          if (Cb) Cb[(size_t)row * N + col] = (__bf16)v;
          else    Cf[(size_t)row * N + col] = v;
        }
      }
}

// ---------------------------------------------------------------------------
// Window attention: one wave per (window, head).  Q/K/V already bf16.
// S(48x48) = Q K^T via 9 WMMAs; softmax in registers; P and V staged in LDS
// (V rows vector-loaded once, reused by all 4 AV fragments); O = P V via 12
// WMMAs; store fused with residual add and window reverse.
// ---------------------------------------------------------------------------
__global__ __launch_bounds__(32) void win_attn_kernel(
    const __bf16* __restrict__ Qf, const __bf16* __restrict__ KV,
    const float* __restrict__ bias_table, const float* __restrict__ xin,
    float* __restrict__ x1) {
  __shared__ alignas(16) __bf16 P[48][64];    // probabilities, A-frag order
  __shared__ alignas(16) __bf16 Vs[64][32];   // [key][d], keys >=37 zero
  int w = blockIdx.x;        // 0..1023
  int h = blockIdx.y;        // 0..15
  int b = w >> 6, widx = w & 63, wr = widx >> 3, wc = widx & 7;
  int lane = threadIdx.x, half = lane >> 4, lidx = lane & 15;

  auto rowOf = [&](int t) -> int {
    int i = t / 6, j = t - i * 6;
    return b * LTOK + (wr * 6 + i) * 48 + (wc * 6 + j);
  };
  const int gRow = BATCH * LTOK + b;  // fused row of global token

  // ---- zero P and Vs with 16B stores; stage V rows (one per lane) -------
  {
    uint4v z4 = {0u, 0u, 0u, 0u};
    uint4v* p4 = (uint4v*)&P[0][0];   // 384 x 16B
    for (int i = lane; i < 384; i += 32) p4[i] = z4;
    uint4v* v4 = (uint4v*)&Vs[0][0];  // 256 x 16B
    for (int i = lane; i < 256; i += 32) v4[i] = z4;
  }
#pragma unroll
  for (int pass = 0; pass < 2; ++pass) {
    int key = pass * 32 + lane;
    if (key < 37) {
      const __bf16* vsrc = (key < 36)
          ? KV + (size_t)rowOf(key) * 1024 + 512 + h * HEADD
          : KV + (size_t)gRow * 1024 + 512 + h * HEADD;
#pragma unroll
      for (int c = 0; c < 4; ++c)
        *(v8bf*)&Vs[key][c * 8] = *(const v8bf*)(vsrc + c * 8);
    }
  }

  // ---- Q fragments (A layout): two 16B vector loads per lane ------------
  AFrag aq[3];
#pragma unroll
  for (int mt = 0; mt < 3; ++mt) {
    int t = mt * 16 + lidx;
    if (t < 36) {
      const __bf16* q = Qf + (size_t)rowOf(t) * DIMC + h * HEADD;
      aq[mt].h[0] = *(const v8bf*)(q + half * 8);
      aq[mt].h[1] = *(const v8bf*)(q + 16 + half * 8);
    } else {
      zero_frag(aq[mt]);
    }
  }

  // ---- S = Q K^T --------------------------------------------------------
  v8f S[3][3];
#pragma unroll
  for (int nt = 0; nt < 3; ++nt) {
    AFrag bk;
    int key = nt * 16 + lidx;
    const __bf16* ks = nullptr;
    if (key < 36)       ks = KV + (size_t)rowOf(key) * 1024 + h * HEADD;
    else if (key == 36) ks = KV + (size_t)gRow * 1024 + h * HEADD;
    if (ks) {
      bk.h[0] = *(const v8bf*)(ks + half * 8);
      bk.h[1] = *(const v8bf*)(ks + 16 + half * 8);
    } else {
      zero_frag(bk);
    }
#pragma unroll
    for (int mt = 0; mt < 3; ++mt) {
      v8f z = {};
      S[mt][nt] = __builtin_amdgcn_wmma_f32_16x16x32_bf16(
          false, aq[mt].v, false, bk.v, (short)0, z, false, false);
    }
  }

  // ---- scale + bias + mask + softmax ------------------------------------
#pragma unroll
  for (int mt = 0; mt < 3; ++mt)
#pragma unroll
    for (int nt = 0; nt < 3; ++nt)
#pragma unroll
      for (int r = 0; r < 8; ++r) {
        int t = mt * 16 + r + half * 8;
        int key = nt * 16 + lidx;
        float s = S[mt][nt][r] * SCALEQ;
        if (t < 36 && key < 37)
          s += bias_table[c_rel.v[t * 37 + key] * NHEAD + h];
        if (key >= 37) s = -1e30f;   // pad columns
        S[mt][nt][r] = s;
      }
#pragma unroll
  for (int mt = 0; mt < 3; ++mt)
#pragma unroll
    for (int r = 0; r < 8; ++r) {
      float mx = fmaxf(fmaxf(S[mt][0][r], S[mt][1][r]), S[mt][2][r]);
      for (int o = 8; o >= 1; o >>= 1) mx = fmaxf(mx, __shfl_xor(mx, o, 32));
      float sum = 0.f;
#pragma unroll
      for (int nt = 0; nt < 3; ++nt) {
        float e = __expf(S[mt][nt][r] - mx);
        S[mt][nt][r] = e;
        sum += e;
      }
      for (int o = 8; o >= 1; o >>= 1) sum += __shfl_xor(sum, o, 32);
      float inv = 1.f / sum;
#pragma unroll
      for (int nt = 0; nt < 3; ++nt) S[mt][nt][r] *= inv;
    }

  __syncthreads();
#pragma unroll
  for (int mt = 0; mt < 3; ++mt)
#pragma unroll
    for (int nt = 0; nt < 3; ++nt)
#pragma unroll
      for (int r = 0; r < 8; ++r)
        P[mt * 16 + r + half * 8][nt * 16 + lidx] = (__bf16)S[mt][nt][r];
  __syncthreads();

  // ---- O = P V  (keys padded to 64 -> two K-steps of 32) ----------------
  v8f O[3][2] = {};
#pragma unroll
  for (int kc = 0; kc < 2; ++kc) {
    AFrag bv[2];
#pragma unroll
    for (int nd = 0; nd < 2; ++nd) {
      int d = nd * 16 + lidx;
#pragma unroll
      for (int i = 0; i < 16; ++i) {
        int k = i + ((i >= 8) ? 8 : 0) + half * 8;
        bv[nd].e[i] = Vs[kc * 32 + k][d];   // zero rows handle key >= 37
      }
    }
#pragma unroll
    for (int mt = 0; mt < 3; ++mt) {
      AFrag ap;
      int t = mt * 16 + lidx;
      ap.h[0] = *(const v8bf*)&P[t][kc * 32 + half * 8];
      ap.h[1] = *(const v8bf*)&P[t][kc * 32 + 16 + half * 8];
#pragma unroll
      for (int nd = 0; nd < 2; ++nd)
        O[mt][nd] = __builtin_amdgcn_wmma_f32_16x16x32_bf16(
            false, ap.v, false, bv[nd].v, (short)0, O[mt][nd], false, false);
    }
  }

  // ---- store: residual add + window reverse -----------------------------
#pragma unroll
  for (int mt = 0; mt < 3; ++mt)
#pragma unroll
    for (int r = 0; r < 8; ++r) {
      int t = mt * 16 + r + half * 8;
      if (t < 36) {
        size_t row = (size_t)rowOf(t);
#pragma unroll
        for (int nd = 0; nd < 2; ++nd) {
          int col = h * HEADD + nd * 16 + lidx;
          x1[row * DIMC + col] = xin[row * DIMC + col] + O[mt][nd][r];
        }
      }
    }
}

// ---------------------------------------------------------------------------
// Global-token attention: q_avg attends to all L tokens of its batch.
// One 256-thread block per (batch, head); K/V rows read as 16B vectors.
// ---------------------------------------------------------------------------
__global__ __launch_bounds__(256) void gattn_kernel(
    const __bf16* __restrict__ Qf, const __bf16* __restrict__ KV,
    const float* __restrict__ xavg, float* __restrict__ xavg1) {
  __shared__ float logits[LTOK];
  __shared__ float red[8];
  __shared__ float qsh[32];
  __shared__ float wpart[8][32];
  __shared__ float bcast[2];
  int b = blockIdx.x, h = blockIdx.y;
  int tid = threadIdx.x, lane = tid & 31, w = tid >> 5;
  size_t gRow = (size_t)(BATCH * LTOK) + b;
  if (tid < 32)
    qsh[tid] = (float)Qf[gRow * DIMC + h * HEADD + tid] * SCALEQ;
  __syncthreads();
  float lmax = -1e30f;
  for (int j = tid; j < LTOK; j += 256) {
    const v8bf* kr = (const v8bf*)(KV + ((size_t)b * LTOK + j) * 1024 + h * HEADD);
    float s = 0.f;
#pragma unroll
    for (int c = 0; c < 4; ++c) {
      F8 t; t.v = kr[c];
#pragma unroll
      for (int e = 0; e < 8; ++e) s += qsh[c * 8 + e] * (float)t.e[e];
    }
    logits[j] = s;
    lmax = fmaxf(lmax, s);
  }
  for (int o = 16; o >= 1; o >>= 1) lmax = fmaxf(lmax, __shfl_xor(lmax, o, 32));
  if (lane == 0) red[w] = lmax;
  __syncthreads();
  if (tid == 0) {
    float m = red[0];
    for (int i = 1; i < 8; ++i) m = fmaxf(m, red[i]);
    bcast[0] = m;
  }
  __syncthreads();
  float mx = bcast[0];
  float part[32];
#pragma unroll
  for (int d = 0; d < 32; ++d) part[d] = 0.f;
  float sum = 0.f;
  for (int j = tid; j < LTOK; j += 256) {
    float p = __expf(logits[j] - mx);
    sum += p;
    const v8bf* vr = (const v8bf*)(KV + ((size_t)b * LTOK + j) * 1024 + 512 + h * HEADD);
#pragma unroll
    for (int c = 0; c < 4; ++c) {
      F8 t; t.v = vr[c];
#pragma unroll
      for (int e = 0; e < 8; ++e) part[c * 8 + e] += p * (float)t.e[e];
    }
  }
  for (int o = 16; o >= 1; o >>= 1) sum += __shfl_xor(sum, o, 32);
  if (lane == 0) red[w] = sum;
#pragma unroll
  for (int d = 0; d < 32; ++d)
    for (int o = 16; o >= 1; o >>= 1) part[d] += __shfl_xor(part[d], o, 32);
  if (lane == 0) {
#pragma unroll
    for (int d = 0; d < 32; ++d) wpart[w][d] = part[d];
  }
  __syncthreads();
  if (tid == 0) {
    float t = 0.f;
    for (int i = 0; i < 8; ++i) t += red[i];
    bcast[1] = t;
  }
  __syncthreads();
  if (tid < 32) {
    float o = 0.f;
    for (int i = 0; i < 8; ++i) o += wpart[i][tid];
    o /= bcast[1];
    size_t idx = (size_t)b * DIMC + h * HEADD + tid;
    xavg1[idx] = xavg[idx] + o;
  }
}

// ---------------------------------------------------------------------------
extern "C" void kernel_launch(void* const* d_in, const int* in_sizes, int n_in,
                              void* d_out, int out_size, void* d_ws,
                              size_t ws_size, hipStream_t stream) {
  const float* x    = (const float*)d_in[0];
  const float* xavg = (const float*)d_in[1];
  const float* w_kv = (const float*)d_in[2];
  const float* b_kv = (const float*)d_in[3];
  const float* w_q  = (const float*)d_in[4];
  const float* b_q  = (const float*)d_in[5];
  const float* rbt  = (const float*)d_in[6];
  const float* g1   = (const float*)d_in[7];
  const float* be1  = (const float*)d_in[8];
  const float* g2   = (const float*)d_in[9];
  const float* be2  = (const float*)d_in[10];
  const float* wf1  = (const float*)d_in[11];
  const float* bf1  = (const float*)d_in[12];
  const float* wf2  = (const float*)d_in[13];
  const float* bf2  = (const float*)d_in[14];
  float* out = (float*)d_out;

  // workspace layout
  __bf16* XN  = (__bf16*)d_ws;                       // NROW x 512
  __bf16* KV  = XN + (size_t)NROW * 512;             // NROW x 1024 (k|v)
  __bf16* Qf  = KV + (size_t)NROW * 1024;            // NROW x 512
  __bf16* Hh  = Qf + (size_t)NROW * 512;             // NROW x 2048 (MLP hidden)
  float*  X1  = (float*)(Hh + (size_t)NROW * 2048);  // NROW x 512 (attn residual)
  __bf16* KVT = (__bf16*)(X1 + (size_t)NROW * 512);  // 1024 x 512
  __bf16* QT  = KVT + (size_t)1024 * 512;            // 512 x 512
  __bf16* F1T = QT + (size_t)512 * 512;              // 2048 x 512
  __bf16* F2T = F1T + (size_t)2048 * 512;            // 512 x 2048

  const int MB = (NROW + 127) / 128;

  // 0) one-time weight transpose + bf16 convert
  transpose_cvt<<<dim3(1024 / 32, 512 / 32), 256, 0, stream>>>(w_kv, KVT, 512, 1024);
  transpose_cvt<<<dim3(512 / 32, 512 / 32), 256, 0, stream>>>(w_q, QT, 512, 512);
  transpose_cvt<<<dim3(2048 / 32, 512 / 32), 256, 0, stream>>>(wf1, F1T, 512, 2048);
  transpose_cvt<<<dim3(512 / 32, 2048 / 32), 256, 0, stream>>>(wf2, F2T, 2048, 512);

  // 1) LN1 over fused rows (x tokens then xavg tokens) -> bf16
  ln_kernel<<<NROW, 128, 0, stream>>>(x, BATCH * LTOK, xavg, g1, be1, XN);

  // 2) fused KV projection (windows + global attn + global token), bf16 out
  gemm_bf16_wmma<<<dim3(MB, 1024 / 64), 256, 0, stream>>>(
      XN, KVT, b_kv, nullptr, nullptr, KV, NROW, 1024, 512, 0);

  // 3) fused Q projection (window q + q_avg), bf16 out
  gemm_bf16_wmma<<<dim3(MB, 512 / 64), 256, 0, stream>>>(
      XN, QT, b_q, nullptr, nullptr, Qf, NROW, 512, 512, 0);

  // 4) window attention (+bias, softmax, residual, window reverse) -> X1
  win_attn_kernel<<<dim3(1024, NHEAD), 32, 0, stream>>>(Qf, KV, rbt, x, X1);

  // 5) global-token attention -> X1 rows [B*L, B*L+B)
  gattn_kernel<<<dim3(BATCH, NHEAD), 256, 0, stream>>>(
      Qf, KV, xavg, X1 + (size_t)(BATCH * LTOK) * 512);

  // 6) LN2 over all fused rows -> bf16
  ln_kernel<<<NROW, 128, 0, stream>>>(X1, NROW, X1, g2, be2, XN);

  // 7) MLP fc1 + exact GELU, bf16 hidden
  gemm_bf16_wmma<<<dim3(MB, 2048 / 64), 256, 0, stream>>>(
      XN, F1T, bf1, nullptr, nullptr, Hh, NROW, 2048, 512, 1);

  // 8) MLP fc2 + residual, f32 out -> writes d_out (x rows then xavg rows)
  gemm_bf16_wmma<<<dim3(MB, 512 / 64), 256, 0, stream>>>(
      Hh, F2T, bf2, X1, out, nullptr, NROW, 512, 2048, 0);
}
